// CrossAttention_78761110274521
// MI455X (gfx1250) — matrix operand order repaired and checked by reading, not compiled, not run on hardware
//
#include <hip/hip_runtime.h>
#include <hip/hip_bf16.h>

typedef __attribute__((ext_vector_type(16))) __bf16 v16bf;
typedef __attribute__((ext_vector_type(8)))  __bf16 v8bf;
typedef __attribute__((ext_vector_type(4)))  __bf16 v4bf;
typedef __attribute__((ext_vector_type(8)))  float  v8f;
typedef int v4i __attribute__((vector_size(16)));   // matches async-LDS builtin param

#define DIMX   768
#define NTOK   197
#define BATCH  128
#define HEADS  12
#define DHEAD  64
#define MROWS  (BATCH * NTOK)   /* 25216 = 197 * 128 */
#define QSCALE 0.125f           /* 64^-0.5 */

#define AS_G __attribute__((address_space(1)))
#define AS_L __attribute__((address_space(3)))

// ---- CDNA5 async global->LDS path (ASYNCcnt), with sync fallback ----------
#if defined(__has_builtin)
#  if __has_builtin(__builtin_amdgcn_global_load_async_to_lds_b128)
#    define HAVE_ASYNC_LDS 1
#  endif
#endif
#ifndef HAVE_ASYNC_LDS
#  define HAVE_ASYNC_LDS 0
#endif

#if HAVE_ASYNC_LDS
#  if __has_builtin(__builtin_amdgcn_s_wait_asynccnt)
#    define WAIT_ASYNC(n) __builtin_amdgcn_s_wait_asynccnt(n)
#  else
#    define WAIT_ASYNC(n) asm volatile("s_wait_asynccnt %0" :: "i"(n) : "memory")
#  endif
#else
#  define WAIT_ASYNC(n) ((void)0)
#endif

// ---------------------------------------------------------------------------
// fp32 -> bf16 conversion (vectorized x4)
// ---------------------------------------------------------------------------
__global__ void cvt_bf16_kernel(const float* __restrict__ in,
                                __bf16* __restrict__ out, int n4) {
    int i = blockIdx.x * blockDim.x + threadIdx.x;
    if (i >= n4) return;
    float4 f = ((const float4*)in)[i];
    v4bf o;
    o[0] = (__bf16)f.x; o[1] = (__bf16)f.y; o[2] = (__bf16)f.z; o[3] = (__bf16)f.w;
    ((v4bf*)out)[i] = o;
}

// W (k-major 768x768 fp32) -> Wt (n-major 768x768 bf16)
__global__ void transpose_cvt_kernel(const float* __restrict__ W,
                                     __bf16* __restrict__ Wt) {
    int idx = blockIdx.x * blockDim.x + threadIdx.x;
    if (idx >= DIMX * DIMX) return;
    int k = idx / DIMX;
    int n = idx - k * DIMX;
    Wt[(size_t)n * DIMX + k] = (__bf16)W[idx];
}

// ---------------------------------------------------------------------------
// bf16 WMMA GEMM: C[M=25216, N=768] = A[M,768] * W[768,768] (+bias) (*scale)
//   A  : bf16 row-major (row, k)
//   Bt : bf16 n-major   (n, k)  == W transposed
// MODE 0: write bf16 head-major  [(b*12+h)*197 + n]*64 + d
// MODE 1: write fp32 row-major   [row*768 + col]
// Block: 256 thr = 8 waves; block tile 128x128; wave tile 64x32; K-step 32.
// Double-buffered LDS; tile fills via GLOBAL_LOAD_ASYNC_TO_LDS when available.
// ---------------------------------------------------------------------------
template <int MODE>
__global__ __launch_bounds__(256)
void wmma_gemm_bf16(const __bf16* __restrict__ A, const __bf16* __restrict__ Bt,
                    const float* __restrict__ bias, float scale,
                    void* __restrict__ Cout) {
    __shared__ __align__(32) __bf16 As[2][128][32];   // [buf][row][k]
    __shared__ __align__(32) __bf16 Bs[2][128][32];   // [buf][n][k]

    const int tid   = threadIdx.x;
    const int lane  = tid & 31;
    const int wave  = tid >> 5;      // 0..7
    const int lrow  = lane & 15;
    const int lh    = lane >> 4;     // half-wave select
    const int waveM = wave & 1;      // 2 waves along M (64 rows each)
    const int waveN = wave >> 1;     // 4 waves along N (32 cols each)

    const int blockM = blockIdx.y * 128;
    const int blockN = blockIdx.x * 128;

    v8f acc[4][2] = {};

    const int arow = tid >> 1;            // 0..127
    const int acol = (tid & 1) * 16;      // 0 or 16

    const __bf16* Ap = A  + (size_t)(blockM + arow) * DIMX + acol;
    const __bf16* Bp = Bt + (size_t)(blockN + arow) * DIMX + acol;

    // Issue the cooperative fill of one 128x32 A tile + 128x32 B tile.
    auto fill = [&](int buf, int kt) {
#if HAVE_ASYNC_LDS
        AS_G char* ga = (AS_G char*)(Ap + kt);
        AS_G char* gb = (AS_G char*)(Bp + kt);
        AS_L char* la = (AS_L char*)&As[buf][arow][acol];
        AS_L char* lb = (AS_L char*)&Bs[buf][arow][acol];
        __builtin_amdgcn_global_load_async_to_lds_b128((AS_G v4i*)ga,        (AS_L v4i*)la,        0, 0);
        __builtin_amdgcn_global_load_async_to_lds_b128((AS_G v4i*)(ga + 16), (AS_L v4i*)(la + 16), 0, 0);
        __builtin_amdgcn_global_load_async_to_lds_b128((AS_G v4i*)gb,        (AS_L v4i*)lb,        0, 0);
        __builtin_amdgcn_global_load_async_to_lds_b128((AS_G v4i*)(gb + 16), (AS_L v4i*)(lb + 16), 0, 0);
#else
        *(v16bf*)&As[buf][arow][acol] = *(const v16bf*)(Ap + kt);
        *(v16bf*)&Bs[buf][arow][acol] = *(const v16bf*)(Bp + kt);
#endif
    };

    fill(0, 0);                       // prologue: 4 async ops in flight

    int buf = 0;
    for (int kt = 0; kt < DIMX; kt += 32, buf ^= 1) {
        if (kt + 32 < DIMX) {
            // Target buffer buf^1 was last read in the *previous* iteration,
            // sealed by the trailing __syncthreads() below -> safe to fill.
            fill(buf ^ 1, kt + 32);
            WAIT_ASYNC(4);            // oldest fill (current buf) complete
        } else {
            WAIT_ASYNC(0);
        }
        __syncthreads();              // current tile visible to all waves

        // B fragment (32x16, KxN): lane holds col n=lrow, k = 16*lh .. +15
        v16bf bfrag[2];
        #pragma unroll
        for (int nt = 0; nt < 2; ++nt)
            bfrag[nt] = *(const v16bf*)&Bs[buf][waveN * 32 + nt * 16 + lrow][16 * lh];

        #pragma unroll
        for (int mt = 0; mt < 4; ++mt) {
            // A fragment (16x32, MxK): lane holds row m=lrow,
            // k = 16*lh .. 16*lh+15
            const int r = waveM * 64 + mt * 16 + lrow;
            v16bf afrag = *(const v16bf*)&As[buf][r][16 * lh];
            #pragma unroll
            for (int nt = 0; nt < 2; ++nt)
                acc[mt][nt] = __builtin_amdgcn_wmma_f32_16x16x32_bf16(
                    false, afrag, false, bfrag[nt], (short)0, acc[mt][nt],
                    false, false);
        }
        __syncthreads();              // seal reads of buf before it is refilled
    }

    // Epilogue: C/D layout — lane holds col=lrow, VGPR v holds row 8*lh+v
    #pragma unroll
    for (int mt = 0; mt < 4; ++mt) {
        #pragma unroll
        for (int nt = 0; nt < 2; ++nt) {
            const int gc = blockN + waveN * 32 + nt * 16 + lrow;
            const float bv = bias[gc];
            #pragma unroll
            for (int v = 0; v < 8; ++v) {
                const int gr = blockM + waveM * 64 + mt * 16 + lh * 8 + v;
                const float val = (acc[mt][nt][v] + bv) * scale;
                if (MODE == 0) {
                    const int b = gr / NTOK;
                    const int n = gr - b * NTOK;
                    const int h = gc >> 6;
                    const int d = gc & 63;
                    ((__bf16*)Cout)[(((size_t)b * HEADS + h) * NTOK + n) * DHEAD + d] =
                        (__bf16)val;
                } else {
                    ((float*)Cout)[(size_t)gr * DIMX + gc] = val;
                }
            }
        }
    }
}

// ---------------------------------------------------------------------------
// CLS-token attention: one 64-thread block per (b,h); full softmax over 197.
// Q already carries the 1/sqrt(d) scale.
// ---------------------------------------------------------------------------
__global__ __launch_bounds__(64)
void attn_cls_kernel(const __bf16* __restrict__ Q, const __bf16* __restrict__ K,
                     const __bf16* __restrict__ V, __bf16* __restrict__ O) {
    const int bh = blockIdx.x;       // b*12 + h
    const int t  = threadIdx.x;      // 0..63
    __shared__ float qs[DHEAD];
    __shared__ float dots[NTOK];

    const __bf16* Qb = Q + (size_t)bh * NTOK * DHEAD;  // token 0
    const __bf16* Kb = K + (size_t)bh * NTOK * DHEAD;
    const __bf16* Vb = V + (size_t)bh * NTOK * DHEAD;

    qs[t] = (float)Qb[t];
    __syncthreads();

    for (int j = t; j < NTOK; j += 64) {
        const __bf16* kj = Kb + (size_t)j * DHEAD;
        float s = 0.f;
        #pragma unroll
        for (int d = 0; d < DHEAD; ++d) s += qs[d] * (float)kj[d];
        dots[j] = s;
    }
    __syncthreads();

    float m = -3.0e38f;
    for (int j = 0; j < NTOK; ++j) m = fmaxf(m, dots[j]);
    __syncthreads();
    for (int j = t; j < NTOK; j += 64) dots[j] = __expf(dots[j] - m);
    __syncthreads();

    float s = 0.f;
    for (int j = 0; j < NTOK; ++j) s += dots[j];
    const float inv = 1.f / s;

    float o = 0.f;
    for (int j = 0; j < NTOK; ++j) o += dots[j] * (float)Vb[(size_t)j * DHEAD + t];

    const int b = bh / HEADS, h = bh - b * HEADS;
    O[((size_t)b * NTOK + 0) * DIMX + h * DHEAD + t] = (__bf16)(o * inv);
}

// ---------------------------------------------------------------------------
// Neighborhood attention for patch tokens 1..196 (device replica of the
// Python _build_neighbors table). Pads to 9 with self-index; padded slots
// are masked to -inf before softmax.
// ---------------------------------------------------------------------------
__device__ inline int build_nb(int i, int nb[9]) {
    int L;
    if (i == 1)        { nb[0]=1;   nb[1]=2;   nb[2]=15;  nb[3]=16;  L=4; }
    else if (i == 14)  { nb[0]=13;  nb[1]=14;  nb[2]=27;  nb[3]=28;  L=4; }
    else if (i == 183) { nb[0]=169; nb[1]=170; nb[2]=183; nb[3]=184; L=4; }
    else if (i == 196) { nb[0]=181; nb[1]=182; nb[2]=195; nb[3]=196; L=4; }
    else if (i >= 2 && i <= 13) {
        nb[0]=i-1;  nb[1]=i;    nb[2]=i+1; nb[3]=i+13; nb[4]=i+14; nb[5]=i+15; L=6;
    } else if (i >= 15 && i <= 169 && ((i - 15) % 14) == 0) {
        nb[0]=i-14; nb[1]=i-13; nb[2]=i;   nb[3]=i+1;  nb[4]=i+14; nb[5]=i+15; L=6;
    } else if (i >= 28 && i <= 182 && ((i - 28) % 14) == 0) {
        nb[0]=i-15; nb[1]=i-14; nb[2]=i-1; nb[3]=i;    nb[4]=i+13; nb[5]=i+14; L=6;
    } else if (i >= 184 && i <= 195) {
        nb[0]=i-15; nb[1]=i-14; nb[2]=i-13; nb[3]=i-1; nb[4]=i;    nb[5]=i+1;  L=6;
    } else {
        nb[0]=i-15; nb[1]=i-14; nb[2]=i-13; nb[3]=i-1; nb[4]=i;    nb[5]=i+1;
        nb[6]=i+13; nb[7]=i+14; nb[8]=i+15; L=9;
    }
    for (int j = L; j < 9; ++j) nb[j] = i;
    return L;
}

__global__ __launch_bounds__(256)
void attn_nbr_kernel(const __bf16* __restrict__ Q, const __bf16* __restrict__ K,
                     const __bf16* __restrict__ V, __bf16* __restrict__ O) {
    const int wave = threadIdx.x >> 5;
    const int lane = threadIdx.x & 31;
    const int bh = blockIdx.x / 25;
    const int pg = blockIdx.x % 25;
    const int p  = pg * 8 + wave;
    if (p >= 196) return;
    const int i = p + 1;

    int nb[9];
    const int L = build_nb(i, nb);

    const __bf16* qr = Q + ((size_t)bh * NTOK + i) * DHEAD + lane * 2;
    const float q0 = (float)qr[0], q1 = (float)qr[1];
    const __bf16* Kb = K + (size_t)bh * NTOK * DHEAD;
    const __bf16* Vb = V + (size_t)bh * NTOK * DHEAD;

    float dots[9];
    float m = -3.0e38f;
    #pragma unroll
    for (int j = 0; j < 9; ++j) {
        const __bf16* kr = Kb + (size_t)nb[j] * DHEAD + lane * 2;
        float ps = q0 * (float)kr[0] + q1 * (float)kr[1];
        #pragma unroll
        for (int off = 16; off > 0; off >>= 1) ps += __shfl_xor(ps, off, 32);
        dots[j] = (j < L) ? ps : -3.0e38f;
        m = fmaxf(m, dots[j]);
    }
    float e[9];
    float s = 0.f;
    #pragma unroll
    for (int j = 0; j < 9; ++j) { e[j] = __expf(dots[j] - m); s += e[j]; }
    const float inv = 1.f / s;

    float o0 = 0.f, o1 = 0.f;
    #pragma unroll
    for (int j = 0; j < 9; ++j) {
        const __bf16* vr = Vb + (size_t)nb[j] * DHEAD + lane * 2;
        o0 += e[j] * (float)vr[0];
        o1 += e[j] * (float)vr[1];
    }

    const int b = bh / HEADS, h = bh - b * HEADS;
    __bf16* op = O + ((size_t)b * NTOK + i) * DIMX + h * DHEAD + lane * 2;
    op[0] = (__bf16)(o0 * inv);
    op[1] = (__bf16)(o1 * inv);
}

// ---------------------------------------------------------------------------
// Launch
// ---------------------------------------------------------------------------
extern "C" void kernel_launch(void* const* d_in, const int* in_sizes, int n_in,
                              void* d_out, int out_size, void* d_ws, size_t ws_size,
                              hipStream_t stream) {
    (void)in_sizes; (void)n_in; (void)out_size; (void)ws_size;

    const float* x_qkv = (const float*)d_in[0];
    const float* y_q   = (const float*)d_in[1];
    const float* Wq    = (const float*)d_in[2];
    const float* bq    = (const float*)d_in[3];
    const float* Wk    = (const float*)d_in[4];
    const float* bk    = (const float*)d_in[5];
    const float* Wv    = (const float*)d_in[6];
    const float* bv    = (const float*)d_in[7];
    const float* Wo    = (const float*)d_in[8];
    const float* bo    = (const float*)d_in[9];

    const size_t S  = (size_t)MROWS * DIMX;   // 19,365,888 elements
    const size_t SW = (size_t)DIMX * DIMX;    // 589,824 elements

    char* ws = (char*)d_ws;
    __bf16* xb  = (__bf16*)ws; ws += S  * 2;
    __bf16* yb  = (__bf16*)ws; ws += S  * 2;
    __bf16* Wqt = (__bf16*)ws; ws += SW * 2;
    __bf16* Wkt = (__bf16*)ws; ws += SW * 2;
    __bf16* Wvt = (__bf16*)ws; ws += SW * 2;
    __bf16* Wot = (__bf16*)ws; ws += SW * 2;
    __bf16* Qh  = (__bf16*)ws; ws += S  * 2;   // head-major, pre-scaled
    __bf16* Kh  = (__bf16*)ws; ws += S  * 2;
    __bf16* Vh  = (__bf16*)ws; ws += S  * 2;
    __bf16* Ob  = (__bf16*)ws; ws += S  * 2;   // attention output, row-major

    // 1) precision conversion
    const int n4 = (int)(S / 4);
    cvt_bf16_kernel<<<(n4 + 255) / 256, 256, 0, stream>>>(x_qkv, xb, n4);
    cvt_bf16_kernel<<<(n4 + 255) / 256, 256, 0, stream>>>(y_q, yb, n4);
    const int nW = DIMX * DIMX;
    transpose_cvt_kernel<<<(nW + 255) / 256, 256, 0, stream>>>(Wq, Wqt);
    transpose_cvt_kernel<<<(nW + 255) / 256, 256, 0, stream>>>(Wk, Wkt);
    transpose_cvt_kernel<<<(nW + 255) / 256, 256, 0, stream>>>(Wv, Wvt);
    transpose_cvt_kernel<<<(nW + 255) / 256, 256, 0, stream>>>(Wo, Wot);

    // 2) projection GEMMs (WMMA bf16, fp32 accumulate)
    dim3 ggrid(DIMX / 128, MROWS / 128);  // (6, 197)
    wmma_gemm_bf16<0><<<ggrid, 256, 0, stream>>>(xb, Wkt, bk, 1.0f,   Kh);
    wmma_gemm_bf16<0><<<ggrid, 256, 0, stream>>>(xb, Wvt, bv, 1.0f,   Vh);
    wmma_gemm_bf16<0><<<ggrid, 256, 0, stream>>>(yb, Wqt, bq, QSCALE, Qh);

    // 3) attention (fp32 softmax)
    attn_cls_kernel<<<BATCH * HEADS, 64, 0, stream>>>(Qh, Kh, Vh, Ob);
    attn_nbr_kernel<<<BATCH * HEADS * 25, 256, 0, stream>>>(Qh, Kh, Vh, Ob);

    // 4) output projection -> fp32 d_out
    wmma_gemm_bf16<1><<<ggrid, 256, 0, stream>>>(Ob, Wot, bo, 1.0f, d_out);
}
